// Phi4FlashDiffAttention_56650618635024
// MI455X (gfx1250) — matrix operand order repaired and checked by reading, not compiled, not run on hardware
//
#include <hip/hip_runtime.h>

typedef __attribute__((ext_vector_type(16))) __bf16        v16bf;
typedef __attribute__((ext_vector_type(8)))  float         v8f;
typedef __attribute__((ext_vector_type(8)))  unsigned int  v8u;

#define SLEN  2048
#define NFULL 16
#define DH    64

__device__ __forceinline__ unsigned int pk_bf16(float a, float b) {
#if __has_builtin(__builtin_amdgcn_cvt_pk_bf16_f32)
  typedef __attribute__((ext_vector_type(2))) __bf16 v2bf;
  v2bf r = __builtin_amdgcn_cvt_pk_bf16_f32(a, b);
  return __builtin_bit_cast(unsigned int, r);
#else
  unsigned int ua = __builtin_bit_cast(unsigned int, a);
  unsigned int ub = __builtin_bit_cast(unsigned int, b);
  ua += 0x7FFFu + ((ua >> 16) & 1u);          // round-to-nearest-even
  ub += 0x7FFFu + ((ub >> 16) & 1u);
  return (ua >> 16) | (ub & 0xFFFF0000u);
#endif
}
__device__ __forceinline__ v8f wmma_bf16(v8u a, v8u b, v8f c) {
  return __builtin_amdgcn_wmma_f32_16x16x32_bf16(
      false, __builtin_bit_cast(v16bf, a),
      false, __builtin_bit_cast(v16bf, b),
      (short)0, c, false, false);
}

// 256 threads = 8 waves per workgroup; workgroup owns 64 query rows of one
// (batch, pair-head). Wave (qt,g): 16-query subtile qt, softmax group g,
// accumulates P_g*V1 and P_g*V2. Double-buffered LDS staging of K1,K2,V1t,V2t
// (f32->bf16) overlaps global loads with the WMMA block. Epilogue: group-1
// waves publish through LDS; group-0 waves do diff-combine + 128-wide RMSNorm.
__global__ __launch_bounds__(256)
void diffattn_kernel(const float* __restrict__ Q, const float* __restrict__ K,
                     const float* __restrict__ V,
                     const float* __restrict__ lq1, const float* __restrict__ lk1,
                     const float* __restrict__ lq2, const float* __restrict__ lk2,
                     const float* __restrict__ W, float* __restrict__ Out)
{
  const int bh   = blockIdx.y;        // 0..15
  const int b    = bh >> 3;
  const int h    = bh & 7;            // pair-head
  const int m0   = blockIdx.x * 64;   // query block base
  const int tid  = threadIdx.x;
  const int wave = tid >> 5;          // 0..7
  const int qt   = wave >> 1;         // query subtile 0..3
  const int g    = wave & 1;          // softmax group 0/1
  const int lane = tid & 31;
  const int half = lane >> 4;
  const int ln   = lane & 15;

  // 33 KB static LDS: staging buffers overlaid with the epilogue exchange.
  __shared__ __align__(16) unsigned char smem[33792];
  unsigned short (*lds_k )[2][32][64] = (unsigned short (*)[2][32][64])smem;           // [buf][grp][key][d]
  unsigned short (*lds_vt)[2][64][32] = (unsigned short (*)[2][64][32])(smem + 16384); // [buf][grp][d][key]
  float (*lds_x)[16][132] = (float (*)[16][132])smem;  // [qt][ln][2*64 + pad] (after loop)

  // lambda_full (uniform)
  float s1 = 0.f, s2 = 0.f;
  for (int i = 0; i < 64; ++i) { s1 += lq1[i] * lk1[i]; s2 += lq2[i] * lk2[i]; }
  const float lam = expf(s1) - expf(s2) + 0.35550906759096926f;   // + LAMBDA_INIT

  // ---- Q fragment for this wave's group (B-layout: lane = query column) ----
  const int mq = m0 + qt * 16 + ln;
  v8u qf[2];                                       // [k-step]
  {
    const float* qp = Q + (((size_t)b * SLEN + mq) * NFULL + (2 * h + g)) * DH;
#pragma unroll
    for (int s = 0; s < 2; ++s) {
      const float4* p4 = (const float4*)(qp + s * 32 + half * 16);
      float4 f0 = p4[0], f1 = p4[1], f2 = p4[2], f3 = p4[3];
      v8u t;
      t[0] = pk_bf16(f0.x, f0.y); t[1] = pk_bf16(f0.z, f0.w);
      t[2] = pk_bf16(f1.x, f1.y); t[3] = pk_bf16(f1.z, f1.w);
      t[4] = pk_bf16(f2.x, f2.y); t[5] = pk_bf16(f2.z, f2.w);
      t[6] = pk_bf16(f3.x, f3.y); t[7] = pk_bf16(f3.z, f3.w);
      qf[s] = t;
    }
  }

  // ---- accumulators: O^T for P_g*V1 (oA) and P_g*V2 (oB) ----
  v8f oA[4], oB[4];
#pragma unroll
  for (int dt = 0; dt < 4; ++dt)
#pragma unroll
    for (int r = 0; r < 8; ++r) { oA[dt][r] = 0.f; oB[dt][r] = 0.f; }

  float mrun = -3.0e38f;
  float lrun = 0.f;
  const float cscale  = 0.18033688011112042f;   // (1/sqrt(64)) * log2(e)
  const int   qbase   = m0 + qt * 16;
  const int   nchunks = (m0 >> 5) + 2;          // >= 2
  const size_t rowstride = (size_t)NFULL * DH;

  // staging helpers (wave-uniform control flow)
  auto stage_loads = [&](int c, float4 kreg[2][2], float4 vreg[2][2]) {
#pragma unroll
    for (int gg = 0; gg < 2; ++gg) {
      const float* kp = K + (((size_t)b * SLEN + c) * NFULL + (2 * h + gg)) * DH;
      const float* vp = V + (((size_t)b * SLEN + c) * NFULL + (2 * h + gg)) * DH;
#pragma unroll
      for (int i = 0; i < 2; ++i) {
        int idx = tid + i * 256;       // 0..511 float4 slots (32 keys x 16)
        int key = idx >> 4;
        int d   = (idx & 15) * 4;
        kreg[gg][i] = *(const float4*)(kp + key * rowstride + d);
        vreg[gg][i] = *(const float4*)(vp + key * rowstride + d);
      }
    }
  };
  auto stage_store = [&](int buf, float4 kreg[2][2], float4 vreg[2][2]) {
#pragma unroll
    for (int gg = 0; gg < 2; ++gg) {
#pragma unroll
      for (int i = 0; i < 2; ++i) {
        int idx = tid + i * 256;
        int key = idx >> 4;
        int d   = (idx & 15) * 4;
        unsigned int* kd = (unsigned int*)&lds_k[buf][gg][key][d];
        kd[0] = pk_bf16(kreg[gg][i].x, kreg[gg][i].y);
        kd[1] = pk_bf16(kreg[gg][i].z, kreg[gg][i].w);
        unsigned int p0 = pk_bf16(vreg[gg][i].x, vreg[gg][i].y);
        unsigned int p1 = pk_bf16(vreg[gg][i].z, vreg[gg][i].w);
        lds_vt[buf][gg][d + 0][key] = (unsigned short)(p0 & 0xFFFFu);
        lds_vt[buf][gg][d + 1][key] = (unsigned short)(p0 >> 16);
        lds_vt[buf][gg][d + 2][key] = (unsigned short)(p1 & 0xFFFFu);
        lds_vt[buf][gg][d + 3][key] = (unsigned short)(p1 >> 16);
      }
    }
  };

  // prologue: fill buffer 0 with chunk 0
  {
    float4 kr[2][2], vr[2][2];
    stage_loads(0, kr, vr);
    stage_store(0, kr, vr);
  }
  __syncthreads();

  for (int ci = 0; ci < nchunks; ++ci) {
    const int  c        = ci * 32;
    const int  cur      = ci & 1;
    const bool havenext = (ci + 1) < nchunks;

    // issue next chunk's global loads (latency hidden behind WMMA block)
    float4 kr[2][2], vr[2][2];
    if (havenext) stage_loads(c + 32, kr, vr);

    const bool need_mask = (c + 31) > qbase;

    // ---- scores S^T = K_g * Q_g^T : two 16-key tiles, 2 k-steps of 32 ----
    v8f sct[2];
#pragma unroll
    for (int t = 0; t < 2; ++t) {
      v8f acc;
#pragma unroll
      for (int r = 0; r < 8; ++r) acc[r] = 0.f;
      const unsigned short* krow = &lds_k[cur][g][t * 16 + ln][0];
#pragma unroll
      for (int s = 0; s < 2; ++s) {
        uint4 r0 = *(const uint4*)(krow + s * 32 + half * 16);
        uint4 r1 = *(const uint4*)(krow + s * 32 + half * 16 + 8);
        v8u a;
        a[0] = r0.x; a[1] = r0.y; a[2] = r0.z; a[3] = r0.w;
        a[4] = r1.x; a[5] = r1.y; a[6] = r1.z; a[7] = r1.w;
        acc = wmma_bf16(a, qf[s], acc);
      }
      sct[t] = acc;
    }
    // ---- causal mask (per-lane select) ----
    if (need_mask) {
#pragma unroll
      for (int t = 0; t < 2; ++t)
#pragma unroll
        for (int r = 0; r < 8; ++r) {
          int key = c + t * 16 + half * 8 + r;
          if (key > mq) sct[t][r] = -3.0e38f;
        }
    }
    // ---- online softmax (own 16 keys + partner lane) ----
    float vmax = -3.0e38f;
#pragma unroll
    for (int t = 0; t < 2; ++t)
#pragma unroll
      for (int r = 0; r < 8; ++r) vmax = fmaxf(vmax, sct[t][r]);
    vmax = fmaxf(vmax, __shfl_xor(vmax, 16));
    const float mnew  = fmaxf(mrun, vmax);
    const float alpha = exp2f((mrun - mnew) * cscale);
    mrun = mnew;
    float p[2][8];
    float psum = 0.f;
#pragma unroll
    for (int t = 0; t < 2; ++t)
#pragma unroll
      for (int r = 0; r < 8; ++r) {
        p[t][r] = exp2f((sct[t][r] - mnew) * cscale);
        psum += p[t][r];
      }
    psum += __shfl_xor(psum, 16);
    lrun = lrun * alpha + psum;

    // ---- assemble P as B-fragment (lane needs keys half*16..+15) ----
    unsigned int own0[4], own1[4];
#pragma unroll
    for (int i = 0; i < 4; ++i) {
      own0[i] = pk_bf16(p[0][2 * i], p[0][2 * i + 1]);   // tile0: keys half*8 + r
      own1[i] = pk_bf16(p[1][2 * i], p[1][2 * i + 1]);   // tile1: keys 16+half*8 + r
    }
    v8u pf;
#pragma unroll
    for (int i = 0; i < 4; ++i) {
      unsigned int send = half ? own0[i] : own1[i];      // what the partner needs
      unsigned int recv = __shfl_xor(send, 16);
      pf[i]     = half ? recv    : own0[i];
      pf[i + 4] = half ? own1[i] : recv;
    }

    // ---- O^T += V^T * P (rescale by alpha first) ----
#pragma unroll
    for (int dt = 0; dt < 4; ++dt) {
      v8f a0 = oA[dt], b0 = oB[dt];
#pragma unroll
      for (int r = 0; r < 8; ++r) { a0[r] *= alpha; b0[r] *= alpha; }
      const unsigned short* vrow0 = &lds_vt[cur][0][dt * 16 + ln][0];
      const unsigned short* vrow1 = &lds_vt[cur][1][dt * 16 + ln][0];
      uint4 u0 = *(const uint4*)(vrow0 + half * 16);
      uint4 u1 = *(const uint4*)(vrow0 + half * 16 + 8);
      v8u va;
      va[0] = u0.x; va[1] = u0.y; va[2] = u0.z; va[3] = u0.w;
      va[4] = u1.x; va[5] = u1.y; va[6] = u1.z; va[7] = u1.w;
      oA[dt] = wmma_bf16(va, pf, a0);
      uint4 w0 = *(const uint4*)(vrow1 + half * 16);
      uint4 w1 = *(const uint4*)(vrow1 + half * 16 + 8);
      v8u vb;
      vb[0] = w0.x; vb[1] = w0.y; vb[2] = w0.z; vb[3] = w0.w;
      vb[4] = w1.x; vb[5] = w1.y; vb[6] = w1.z; vb[7] = w1.w;
      oB[dt] = wmma_bf16(vb, pf, b0);
    }

    if (havenext) stage_store(cur ^ 1, kr, vr);
    __syncthreads();
  }

  // ---- epilogue ----
  const float li = 1.0f / lrun;

  if (g) {
    // group-1 waves publish O21/l2 and O22/l2 via LDS (staging LDS is dead now)
#pragma unroll
    for (int dt = 0; dt < 4; ++dt) {
      const int d0 = dt * 16 + half * 8;
      float4 x0, x1, y0, y1;
      x0.x = oA[dt][0] * li; x0.y = oA[dt][1] * li; x0.z = oA[dt][2] * li; x0.w = oA[dt][3] * li;
      x1.x = oA[dt][4] * li; x1.y = oA[dt][5] * li; x1.z = oA[dt][6] * li; x1.w = oA[dt][7] * li;
      y0.x = oB[dt][0] * li; y0.y = oB[dt][1] * li; y0.z = oB[dt][2] * li; y0.w = oB[dt][3] * li;
      y1.x = oB[dt][4] * li; y1.y = oB[dt][5] * li; y1.z = oB[dt][6] * li; y1.w = oB[dt][7] * li;
      *(float4*)&lds_x[qt][ln][d0]          = x0;
      *(float4*)&lds_x[qt][ln][d0 + 4]      = x1;
      *(float4*)&lds_x[qt][ln][64 + d0]     = y0;
      *(float4*)&lds_x[qt][ln][64 + d0 + 4] = y1;
    }
  }
  __syncthreads();
  if (!g) {
    // group-0 waves: diff-combine, 128-wide RMSNorm, weighted store
    float ssq = 0.f;
#pragma unroll
    for (int dt = 0; dt < 4; ++dt) {
      const int d0 = dt * 16 + half * 8;
      float4 p0 = *(const float4*)&lds_x[qt][ln][d0];
      float4 p1 = *(const float4*)&lds_x[qt][ln][d0 + 4];
      float4 q0 = *(const float4*)&lds_x[qt][ln][64 + d0];
      float4 q1 = *(const float4*)&lds_x[qt][ln][64 + d0 + 4];
      float pa[8] = { p0.x, p0.y, p0.z, p0.w, p1.x, p1.y, p1.z, p1.w };
      float qa[8] = { q0.x, q0.y, q0.z, q0.w, q1.x, q1.y, q1.z, q1.w };
#pragma unroll
      for (int r = 0; r < 8; ++r) {
        float d1 = oA[dt][r] * li - lam * pa[r];
        float d2 = oB[dt][r] * li - lam * qa[r];
        oA[dt][r] = d1; oB[dt][r] = d2;
        ssq += d1 * d1 + d2 * d2;
      }
    }
    ssq += __shfl_xor(ssq, 16);
    const float rfac = rsqrtf(ssq * (1.0f / 128.0f) + 1e-5f) * 0.6444909324090307f; // *(1-LAMBDA_INIT)

    float* op = Out + (((size_t)b * SLEN + mq) * NFULL + 2 * h) * DH;
#pragma unroll
    for (int dt = 0; dt < 4; ++dt) {
      const int d0 = dt * 16 + half * 8;
      float4 w0 = *(const float4*)(W + d0);
      float4 w1 = *(const float4*)(W + d0 + 4);
      float4 x0, x1;
      x0.x = oA[dt][0] * rfac * w0.x; x0.y = oA[dt][1] * rfac * w0.y;
      x0.z = oA[dt][2] * rfac * w0.z; x0.w = oA[dt][3] * rfac * w0.w;
      x1.x = oA[dt][4] * rfac * w1.x; x1.y = oA[dt][5] * rfac * w1.y;
      x1.z = oA[dt][6] * rfac * w1.z; x1.w = oA[dt][7] * rfac * w1.w;
      *(float4*)(op + d0)     = x0;
      *(float4*)(op + d0 + 4) = x1;
      float4 w2 = *(const float4*)(W + 64 + d0);
      float4 w3 = *(const float4*)(W + 64 + d0 + 4);
      float4 y0, y1;
      y0.x = oB[dt][0] * rfac * w2.x; y0.y = oB[dt][1] * rfac * w2.y;
      y0.z = oB[dt][2] * rfac * w2.z; y0.w = oB[dt][3] * rfac * w2.w;
      y1.x = oB[dt][4] * rfac * w3.x; y1.y = oB[dt][5] * rfac * w3.y;
      y1.z = oB[dt][6] * rfac * w3.z; y1.w = oB[dt][7] * rfac * w3.w;
      *(float4*)(op + DH + d0)     = y0;
      *(float4*)(op + DH + d0 + 4) = y1;
    }
  }
}

extern "C" void kernel_launch(void* const* d_in, const int* in_sizes, int n_in,
                              void* d_out, int out_size, void* d_ws, size_t ws_size,
                              hipStream_t stream) {
  (void)in_sizes; (void)n_in; (void)out_size; (void)d_ws; (void)ws_size;
  const float* q   = (const float*)d_in[0];
  const float* k   = (const float*)d_in[1];
  const float* v   = (const float*)d_in[2];
  const float* lq1 = (const float*)d_in[3];
  const float* lk1 = (const float*)d_in[4];
  const float* lq2 = (const float*)d_in[5];
  const float* lk2 = (const float*)d_in[6];
  const float* w   = (const float*)d_in[7];
  float* out = (float*)d_out;

  dim3 grid(SLEN / 64, 2 * 8);   // (query blocks, B * pair-heads)
  diffattn_kernel<<<grid, 256, 0, stream>>>(q, k, v, lq1, lk1, lq2, lk2, w, out);
}
